// SpectralWaveletGateBlock_77240691851288
// MI455X (gfx1250) — matrix operand order repaired and checked by reading, not compiled, not run on hardware
//
#include <hip/hip_runtime.h>
#include <math.h>

typedef _Float16 h4  __attribute__((ext_vector_type(4)));
typedef _Float16 h8  __attribute__((ext_vector_type(8)));
typedef _Float16 h16 __attribute__((ext_vector_type(16)));
typedef float    f8  __attribute__((ext_vector_type(8)));

#define RES_ 128
#define C_   256
#define B_   4
#define HW_  (RES_*RES_)
#define LDF_ 192   // 96 real | 96 imag per spectral row (65 valid, padded to 96)
#define HF_  96

// ---------------- workspace layout (byte offsets) ----------------
#define O_X2H  ((size_t)0)                         // (B*C, HW) f16          32MB
#define O_XH   (O_X2H + (size_t)B_*C_*HW_*2)       // (B*HW, C) f16          32MB
#define O_S1   (O_XH  + (size_t)B_*HW_*C_*2)       // XrowRI -> Xfused f16   48MB
#define O_S2   (O_S1  + (size_t)B_*C_*RES_*LDF_*2) // Xf -> xspec f16        48MB
#define O_S3   (O_S2  + (size_t)B_*C_*RES_*LDF_*2) // Xmap -> Z f16          48MB
#define O_F1   (O_S3  + (size_t)B_*C_*RES_*LDF_*2) // ssum f32               64MB
#define O_F2   (O_F1  + (size_t)B_*C_*HW_*4)       // lin f32                64MB
#define O_F3   (O_F2  + (size_t)B_*HW_*C_*4)       // pre f32                64MB
#define O_BFWD (O_F3  + (size_t)B_*HW_*C_*4)       // BfwdT (192x128) f16
#define O_BINV (O_BFWD + (size_t)192*128*2)        // BinvT (128x192) f16
#define O_DRH  (O_BINV + (size_t)128*192*2)        // DrH (128x128) f16
#define O_DIH  (O_DRH  + (size_t)128*128*2)
#define O_WRH  (O_DIH  + (size_t)128*128*2)        // WrH (256x256) f16
#define O_WIH  (O_WRH  + (size_t)C_*C_*2)
#define O_LNT  (O_WIH  + (size_t)C_*C_*2)          // linT (256x256) f16 (already N x K)
#define O_CNT  (O_LNT  + (size_t)C_*C_*2)          // band counts f32
#define O_MEAN (O_CNT  + 256)
#define O_ALPH (O_MEAN + (size_t)B_*C_*8*4)

// ---------------- CDNA5 async-copy helpers (inline asm) ----------------
__device__ __forceinline__ void async_b128(unsigned int lds, const void* gp) {
  unsigned long long g = (unsigned long long)gp;
  asm volatile("global_load_async_to_lds_b128 %0, %1, off"
               :: "v"(lds), "v"(g) : "memory");
}
__device__ __forceinline__ void wait_async0() {
  asm volatile("s_wait_asynccnt 0x0" ::: "memory");
}
__device__ __forceinline__ void wait_async2() {
  asm volatile("s_wait_asynccnt 0x2" ::: "memory");
}
__device__ __forceinline__ void wait_async4() {
  asm volatile("s_wait_asynccnt 0x4" ::: "memory");
}
__device__ __forceinline__ unsigned int lds_addr(const void* p) {
  return (unsigned int)(size_t)p;   // LDS generic address: low 32 bits = LDS offset
}

// ---------------- WMMA helpers ----------------
__device__ __forceinline__ f8 f8_zero() {
  f8 z;
#pragma unroll
  for (int i = 0; i < 8; ++i) z[i] = 0.f;
  return z;
}

__device__ __forceinline__ f8 wmma_f16(h16 a, h16 b, f8 c) {
  return __builtin_amdgcn_wmma_f32_16x16x32_f16(false, a, false, b, (short)0, c, false, false);
}

__device__ __forceinline__ h16 cat_h16(h8 lo, h8 hi) {
  h16 r;
#pragma unroll
  for (int i = 0; i < 8; ++i) { r[i] = lo[i]; r[8 + i] = hi[i]; }
  return r;
}

// A fragment from LDS tile stored [row][32] halves (row-major, K contiguous)
// ISA 16-bit A layout: row = lane%16 ; K = 16*(e>>3) + 8*(lane/16) + (e&7)
__device__ __forceinline__ h16 lds_afrag(const _Float16* t, int row0, int lane) {
  const _Float16* p = t + (size_t)(row0 + (lane & 15)) * 32;
  int h = lane >> 4;
  h8 lo = *(const h8*)(p + 8 * h);
  h8 hi = *(const h8*)(p + 16 + 8 * h);
  return cat_h16(lo, hi);
}

// B fragment from LDS tile stored transposed [n][32] halves
// ISA 16-bit B layout: col n = lane%16 ; K = 16*(e>>3) + 8*(lane/16) + (e&7)
__device__ __forceinline__ h16 lds_bfrag(const _Float16* t, int n0, int lane) {
  const _Float16* p = t + (size_t)(n0 + (lane & 15)) * 32;
  h8 lo = *(const h8*)(p + 8 * (lane >> 4));
  h8 hi = *(const h8*)(p + 16 + 8 * (lane >> 4));
  return cat_h16(lo, hi);
}

__device__ __forceinline__ int band_id(int m, int k) {
  float yy = (float)m * (1.0f / 127.0f);
  float xx = (float)k * (1.0f / 64.0f);
  float r  = sqrtf(yy * yy + xx * xx);
  const float rmax = 1.41421356237309515f + 1e-8f;
  int b = (int)floorf(r / rmax * 6.0f);
  return b > 5 ? 5 : b;
}

__device__ __forceinline__ float sigm(float v) { return 1.f / (1.f + expf(-v)); }

// ---------------- K0: constant init (f16 DFT matrices, pre-transposed B) ---------
__global__ void k_init_consts(_Float16* __restrict__ BfwdT, _Float16* __restrict__ BinvT,
                              _Float16* __restrict__ DrH, _Float16* __restrict__ DiH) {
  const float invs = 0.0883883476483184406f;          // 1/sqrt(128)
  const float step = 6.28318530717958648f / 128.f;
  int tid = blockIdx.x * blockDim.x + threadIdx.x;
  int stride = gridDim.x * blockDim.x;
  // BfwdT[n][w]: n<96 -> real coeff of freq k=n ; n>=96 -> imag coeff of k=n-96
  for (int idx = tid; idx < 192 * 128; idx += stride) {
    int n = idx / 128, w = idx % 128;
    int k = n % HF_;
    float v = 0.f;
    if (k < 65) {
      float ang = step * (float)((w * k) & 127);
      v = (n < HF_) ? cosf(ang) * invs : -sinf(ang) * invs;
    }
    BfwdT[idx] = (_Float16)v;
  }
  // column DFT (ortho): D[m][r] = e^{-2pi i m r/128}/sqrt(128)
  for (int idx = tid; idx < 128 * 128; idx += stride) {
    int m = idx / 128, r = idx % 128;
    float ang = step * (float)((m * r) & 127);
    DrH[idx] = (_Float16)(cosf(ang) * invs);
    DiH[idx] = (_Float16)(-sinf(ang) * invs);
  }
  // BinvT[n][kr]: n = output w, kr in [0,192): Hermitian fold weights
  for (int idx = tid; idx < 128 * 192; idx += stride) {
    int n = idx / 192, kr = idx % 192;
    int k = kr % HF_;
    float v = 0.f;
    if (k < 65) {
      float wgt = (k == 0 || k == 64) ? 1.f : 2.f;
      float ang = step * (float)((k * n) & 127);
      v = (kr < HF_) ? wgt * cosf(ang) * invs : -wgt * sinf(ang) * invs;
    }
    BinvT[idx] = (_Float16)v;
  }
}

__global__ __launch_bounds__(256) void k_init_counts(float* __restrict__ counts) {
  __shared__ float s[6 * 256];
  float loc[6] = {0, 0, 0, 0, 0, 0};
  for (int f = threadIdx.x; f < 128 * 65; f += 256)
    loc[band_id(f / 65, f % 65)] += 1.f;
  for (int b = 0; b < 6; ++b) s[b * 256 + threadIdx.x] = loc[b];
  __syncthreads();
  for (int off = 128; off > 0; off >>= 1) {
    if ((int)threadIdx.x < off)
      for (int b = 0; b < 6; ++b) s[b * 256 + threadIdx.x] += s[b * 256 + threadIdx.x + off];
    __syncthreads();
  }
  if (threadIdx.x < 6) counts[threadIdx.x] = fmaxf(s[threadIdx.x * 256], 1.f);
}

// ---------------- f32 -> f16 convert ----------------
__global__ void k_f32_to_f16(const float* __restrict__ src, _Float16* __restrict__ dst, int n) {
  int i = blockIdx.x * blockDim.x + threadIdx.x;
  int stride = gridDim.x * blockDim.x;
  for (; i < n; i += stride) dst[i] = (_Float16)src[i];
}

// ---------------- K1: transpose x (B,HW,C) f32 -> x2h (B*C, HW) f16 ----------------
__global__ __launch_bounds__(256) void k_transpose_in(const float* __restrict__ x,
                                                      _Float16* __restrict__ x2h) {
  __shared__ float t[32][33];
  int b = blockIdx.z;
  int c0 = blockIdx.x * 32, p0 = blockIdx.y * 32;
  int tx = threadIdx.x, ty = threadIdx.y;
#pragma unroll
  for (int i = 0; i < 4; ++i) {
    int p = p0 + ty + i * 8;
    t[ty + i * 8][tx] = x[((size_t)b * HW_ + p) * C_ + c0 + tx];
  }
  __syncthreads();
#pragma unroll
  for (int i = 0; i < 4; ++i) {
    int c = c0 + ty + i * 8;
    x2h[((size_t)(b * C_ + c)) * HW_ + p0 + tx] = (_Float16)t[tx][ty + i * 8];
  }
}

// ---------------- f16 WMMA GEMM, async double-buffered LDS ----------------
// C = A(MxK,f16) * BT(NxK,f16)^T [+bias]. Grid tiles M,N exactly; K%32==0.
__device__ __forceinline__ void gemm_issue(const _Float16* A, int lda,
                                           const _Float16* BT, int ldb,
                                           int mb, int nb, int k0, int tid,
                                           _Float16* At, _Float16* Bt) {
#pragma unroll
  for (int i = 0; i < 2; ++i) {
    int q = tid + i * 128;
    int r = q >> 2, c8 = (q & 3) * 8;
    async_b128(lds_addr(At + r * 32 + c8), A + (size_t)(mb + r) * lda + k0 + c8);
  }
#pragma unroll
  for (int i = 0; i < 2; ++i) {
    int q = tid + i * 128;
    int n = q >> 2, c8 = (q & 3) * 8;
    async_b128(lds_addr(Bt + n * 32 + c8), BT + (size_t)(nb + n) * ldb + k0 + c8);
  }
}

template <typename OutT>
__global__ __launch_bounds__(128) void k_gemm(
    const _Float16* __restrict__ A, int lda,
    const _Float16* __restrict__ BT, int ldb,
    const float* __restrict__ bias,
    OutT* __restrict__ Cmat, int ldc, int K) {
  __shared__ alignas(16) _Float16 At[2][64 * 32];
  __shared__ alignas(16) _Float16 Bt[2][64 * 32];
  int tid = threadIdx.x, lane = tid & 31, wave = tid >> 5;
  int mb = blockIdx.y * 64, nb = blockIdx.x * 64;
  f8 acc[4];
#pragma unroll
  for (int t = 0; t < 4; ++t) acc[t] = f8_zero();

  int S = K >> 5;
  gemm_issue(A, lda, BT, ldb, mb, nb, 0, tid, At[0], Bt[0]);
  for (int s = 0; s < S; ++s) {
    int p = s & 1;
    if (s + 1 < S) {
      gemm_issue(A, lda, BT, ldb, mb, nb, (s + 1) << 5, tid, At[p ^ 1], Bt[p ^ 1]);
      wait_async4();  // stage s complete; stage s+1 still in flight
    } else {
      wait_async0();
    }
    __syncthreads();
    h16 a = lds_afrag(At[p], wave * 16, lane);
#pragma unroll
    for (int t = 0; t < 4; ++t) {
      h16 b = lds_bfrag(Bt[p], t * 16, lane);
      acc[t] = wmma_f16(a, b, acc[t]);
    }
    __syncthreads();
  }
  int r0 = wave * 16 + (lane >> 4) * 8;
  int cc0 = lane & 15;
  if (sizeof(OutT) == 2) {
    // bounce through LDS, write contiguous b128 chunks
    _Float16* Ct = &At[0][0];  // 4096 halves = 64x64 tile
#pragma unroll
    for (int t = 0; t < 4; ++t) {
      float bv = bias ? bias[nb + cc0 + t * 16] : 0.f;
#pragma unroll
      for (int i = 0; i < 8; ++i)
        Ct[(size_t)(r0 + i) * 64 + t * 16 + cc0] = (_Float16)(acc[t][i] + bv);
    }
    __syncthreads();
    _Float16* Ch = reinterpret_cast<_Float16*>(Cmat);
#pragma unroll
    for (int i = 0; i < 4; ++i) {
      int q = tid + i * 128;
      int row = q >> 3, c8 = (q & 7) * 8;
      *(h8*)(Ch + (size_t)(mb + row) * ldc + nb + c8) = *(const h8*)(Ct + row * 64 + c8);
    }
  } else {
#pragma unroll
    for (int t = 0; t < 4; ++t) {
      int gc = nb + cc0 + t * 16;
      float bv = bias ? bias[gc] : 0.f;
#pragma unroll
      for (int i = 0; i < 8; ++i)
        Cmat[(size_t)(mb + r0 + i) * ldc + gc] = (OutT)(acc[t][i] + bv);
    }
  }
}

// ---------------- batched complex left-DFT: Y = D (*) X per image (f16) ----------
// sign=+1: Yr=rr-ii, Yi=ri+ir ; sign=-1 (conj D): Yr=rr+ii, Yi=ri-ir
__global__ __launch_bounds__(256) void k_coldft(
    const _Float16* __restrict__ DrH, const _Float16* __restrict__ DiH,
    const _Float16* __restrict__ X, _Float16* __restrict__ Y, int sign) {
  __shared__ alignas(16) _Float16 Art[2][64 * 32], Ait[2][64 * 32];   // 16KB
  __shared__ alignas(16) _Float16 Brt[2][32 * 32], Bit[2][32 * 32];   // 8KB
  int tid = threadIdx.x, lane = tid & 31, wave = tid >> 5;
  int wm = wave & 3, wn = wave >> 2;
  int mb = blockIdx.y * 64, nb = blockIdx.x * 32;
  int img = blockIdx.z;
  const _Float16* Xb = X + (size_t)img * RES_ * LDF_;
  f8 arr = f8_zero(), aii = f8_zero(), ari = f8_zero(), air = f8_zero();

  int ra = tid >> 2, c8a = (tid & 3) * 8;   // A chunk coords
  int kk = tid >> 3, nq = (tid & 7) * 4;    // B quad coords
  int gn = nb + nq;

  // prologue: B regs for stage 0, async A for stage 0
  h4 br_ = *(const h4*)(Xb + (size_t)kk * LDF_ + gn);
  h4 bi_ = *(const h4*)(Xb + (size_t)kk * LDF_ + HF_ + gn);
  async_b128(lds_addr(Art[0] + ra * 32 + c8a), DrH + (size_t)(mb + ra) * 128 + c8a);
  async_b128(lds_addr(Ait[0] + ra * 32 + c8a), DiH + (size_t)(mb + ra) * 128 + c8a);

  for (int s = 0; s < 4; ++s) {
    int p = s & 1;
#pragma unroll
    for (int j = 0; j < 4; ++j) {   // commit B regs
      Brt[p][(nq + j) * 32 + kk] = br_[j];
      Bit[p][(nq + j) * 32 + kk] = bi_[j];
    }
    if (s < 3) {
      int k1 = (s + 1) * 32;
      async_b128(lds_addr(Art[p ^ 1] + ra * 32 + c8a), DrH + (size_t)(mb + ra) * 128 + k1 + c8a);
      async_b128(lds_addr(Ait[p ^ 1] + ra * 32 + c8a), DiH + (size_t)(mb + ra) * 128 + k1 + c8a);
      wait_async2();
    } else {
      wait_async0();
    }
    __syncthreads();
    if (s < 3) {  // software-pipelined B loads for next stage
      int k1 = (s + 1) * 32;
      br_ = *(const h4*)(Xb + (size_t)(k1 + kk) * LDF_ + gn);
      bi_ = *(const h4*)(Xb + (size_t)(k1 + kk) * LDF_ + HF_ + gn);
    }
    h16 ar = lds_afrag(Art[p], wm * 16, lane);
    h16 ai = lds_afrag(Ait[p], wm * 16, lane);
    h16 br = lds_bfrag(Brt[p], wn * 16, lane);
    h16 bi = lds_bfrag(Bit[p], wn * 16, lane);
    arr = wmma_f16(ar, br, arr);
    aii = wmma_f16(ai, bi, aii);
    ari = wmma_f16(ar, bi, ari);
    air = wmma_f16(ai, br, air);
    __syncthreads();
  }
  // epilogue bounce through LDS (reuse Art storage: 4096 halves = 64x32 x {re,im})
  float sg = (float)sign;
  _Float16* Ct = &Art[0][0];
  int r0 = wm * 16 + (lane >> 4) * 8;
  int cc = wn * 16 + (lane & 15);
#pragma unroll
  for (int i = 0; i < 8; ++i) {
    Ct[(size_t)(r0 + i) * 32 + cc]        = (_Float16)(arr[i] - sg * aii[i]);
    Ct[2048 + (size_t)(r0 + i) * 32 + cc] = (_Float16)(ari[i] + sg * air[i]);
  }
  __syncthreads();
  _Float16* Yb = Y + (size_t)img * RES_ * LDF_;
#pragma unroll
  for (int i = 0; i < 2; ++i) {
    int q = tid + i * 256;
    int part = q >> 8, rem = q & 255;
    int row = rem >> 2, c8 = (rem & 3) * 8;
    *(h8*)(Yb + (size_t)(mb + row) * LDF_ + part * HF_ + nb + c8) =
        *(const h8*)(Ct + part * 2048 + row * 32 + c8);
  }
}

// ---------------- band reduce: means (B*C, 6) of |Xf| per radial band -------------
__global__ __launch_bounds__(256) void k_band_reduce(
    const _Float16* __restrict__ Xf, const float* __restrict__ counts,
    float* __restrict__ means) {
  int bc = blockIdx.x;
  __shared__ float s[6 * 256];
  float loc[6] = {0, 0, 0, 0, 0, 0};
  for (int f = threadIdx.x; f < 128 * 65; f += 256) {
    int m = f / 65, k = f % 65;
    size_t base = ((size_t)bc * RES_ + m) * LDF_ + k;
    float xr = (float)Xf[base], xi = (float)Xf[base + HF_];
    loc[band_id(m, k)] += sqrtf(xr * xr + xi * xi);
  }
  for (int b = 0; b < 6; ++b) s[b * 256 + threadIdx.x] = loc[b];
  __syncthreads();
  for (int off = 128; off > 0; off >>= 1) {
    if ((int)threadIdx.x < off)
      for (int b = 0; b < 6; ++b) s[b * 256 + threadIdx.x] += s[b * 256 + threadIdx.x + off];
    __syncthreads();
  }
  if (threadIdx.x < 6)
    means[bc * 6 + threadIdx.x] = s[threadIdx.x * 256] / (counts[threadIdx.x] + 1e-6f);
}

// ---------------- band MLP: 6 -> 128 relu -> 6 sigmoid ----------------
__global__ __launch_bounds__(128) void k_band_mlp(
    const float* __restrict__ means,
    const float* __restrict__ w1, const float* __restrict__ b1,
    const float* __restrict__ w2, const float* __restrict__ b2,
    float* __restrict__ alpha) {
  int bc = blockIdx.x;
  __shared__ float hm[128];
  __shared__ float mloc[6];
  if (threadIdx.x < 6) mloc[threadIdx.x] = means[bc * 6 + threadIdx.x];
  __syncthreads();
  float a = b1[threadIdx.x];
#pragma unroll
  for (int n = 0; n < 6; ++n) a += w1[threadIdx.x * 6 + n] * mloc[n];
  hm[threadIdx.x] = a > 0.f ? a : 0.f;
  __syncthreads();
  if (threadIdx.x < 6) {
    float acc = b2[threadIdx.x];
    for (int j = 0; j < 128; ++j) acc += w2[threadIdx.x * 128 + j] * hm[j];
    alpha[bc * 6 + threadIdx.x] = sigm(acc);
  }
}

// ---------------- channel mix: Xmap = Wc (*) (mask . Xf) (complex, per batch b) ---
__global__ __launch_bounds__(256) void k_chanmix(
    const _Float16* __restrict__ WrH, const _Float16* __restrict__ WiH,
    const _Float16* __restrict__ Xf,
    const float* __restrict__ kx, const float* __restrict__ ky,
    _Float16* __restrict__ Xmap) {
  __shared__ alignas(16) _Float16 Art[2][64 * 32], Ait[2][64 * 32];
  __shared__ alignas(16) _Float16 Brt[2][32 * 32], Bit[2][32 * 32];
  int tid = threadIdx.x, lane = tid & 31, wave = tid >> 5;
  int wm = wave & 3, wn = wave >> 2;
  int mb = blockIdx.y * 64;   // channel tile
  int nb = blockIdx.x * 32;   // over 128*96 frequencies
  int b = blockIdx.z;
  int kxi = (int)floorf(sigm(kx[0]) * 128.f);
  int kyi = (int)floorf(sigm(ky[0]) * 65.f);
  f8 arr = f8_zero(), aii = f8_zero(), ari = f8_zero(), air = f8_zero();

  int ra = tid >> 2, c8a = (tid & 3) * 8;
  int kk = tid >> 3, nq = (tid & 7) * 4;
  int n = nb + nq;                   // quads never straddle the 96 boundary (96%4==0)
  int mf = n / HF_, kc = n % HF_;

  // prologue
  size_t base0 = ((size_t)(b * C_ + kk) * RES_ + mf) * LDF_ + kc;
  h4 br_ = *(const h4*)(Xf + base0);
  h4 bi_ = *(const h4*)(Xf + base0 + HF_);
  async_b128(lds_addr(Art[0] + ra * 32 + c8a), WrH + (size_t)(mb + ra) * C_ + c8a);
  async_b128(lds_addr(Ait[0] + ra * 32 + c8a), WiH + (size_t)(mb + ra) * C_ + c8a);

  for (int s = 0; s < 8; ++s) {
    int p = s & 1;
#pragma unroll
    for (int j = 0; j < 4; ++j) {   // commit masked B regs
      bool on = (mf < kxi) && ((kc + j) < kyi) && ((kc + j) < 65);
      Brt[p][(nq + j) * 32 + kk] = on ? br_[j] : (_Float16)0.f;
      Bit[p][(nq + j) * 32 + kk] = on ? bi_[j] : (_Float16)0.f;
    }
    if (s < 7) {
      int k1 = (s + 1) * 32;
      async_b128(lds_addr(Art[p ^ 1] + ra * 32 + c8a), WrH + (size_t)(mb + ra) * C_ + k1 + c8a);
      async_b128(lds_addr(Ait[p ^ 1] + ra * 32 + c8a), WiH + (size_t)(mb + ra) * C_ + k1 + c8a);
      wait_async2();
    } else {
      wait_async0();
    }
    __syncthreads();
    if (s < 7) {
      int k1 = (s + 1) * 32;
      size_t base = ((size_t)(b * C_ + k1 + kk) * RES_ + mf) * LDF_ + kc;
      br_ = *(const h4*)(Xf + base);
      bi_ = *(const h4*)(Xf + base + HF_);
    }
    h16 ar = lds_afrag(Art[p], wm * 16, lane);
    h16 ai = lds_afrag(Ait[p], wm * 16, lane);
    h16 br = lds_bfrag(Brt[p], wn * 16, lane);
    h16 bi = lds_bfrag(Bit[p], wn * 16, lane);
    arr = wmma_f16(ar, br, arr);
    aii = wmma_f16(ai, bi, aii);
    ari = wmma_f16(ar, bi, ari);
    air = wmma_f16(ai, br, air);
    __syncthreads();
  }
  // epilogue bounce: rows are channels, cols are 32 contiguous freqs within one m-row
  _Float16* Ct = &Art[0][0];
  int r0 = wm * 16 + (lane >> 4) * 8;
  int cc = wn * 16 + (lane & 15);
#pragma unroll
  for (int i = 0; i < 8; ++i) {
    Ct[(size_t)(r0 + i) * 32 + cc]        = (_Float16)(arr[i] - aii[i]);
    Ct[2048 + (size_t)(r0 + i) * 32 + cc] = (_Float16)(ari[i] + air[i]);
  }
  __syncthreads();
  int mrow = nb / HF_, kc0 = nb % HF_;
#pragma unroll
  for (int i = 0; i < 2; ++i) {
    int q = tid + i * 256;
    int part = q >> 8, rem = q & 255;
    int row = rem >> 2, c8 = (rem & 3) * 8;
    *(h8*)(Xmap + ((size_t)(b * C_ + mb + row) * RES_ + mrow) * LDF_ + part * HF_ + kc0 + c8) =
        *(const h8*)(Ct + part * 2048 + row * 32 + c8);
  }
}

// ---------------- fuse: alpha*Xmap (low) + (1-alpha)*Xf (high) ----------------
__global__ __launch_bounds__(256) void k_fuse(
    const _Float16* __restrict__ Xf, const _Float16* __restrict__ Xmap,
    const float* __restrict__ alpha,
    const float* __restrict__ kx, const float* __restrict__ ky,
    _Float16* __restrict__ Xfused) {
  int bc = blockIdx.x;
  int kxi = (int)floorf(sigm(kx[0]) * 128.f);
  int kyi = (int)floorf(sigm(ky[0]) * 65.f);
  const float* ab = alpha + bc * 6;
  for (int idx = threadIdx.x; idx < 128 * HF_; idx += 256) {
    int m = idx / HF_, k = idx % HF_;
    size_t base = ((size_t)bc * RES_ + m) * LDF_ + k;
    float fr = 0.f, fi = 0.f;
    if (k < 65) {
      float a = ab[band_id(m, k)];
      if (m < kxi && k < kyi) { fr = a * (float)Xmap[base]; fi = a * (float)Xmap[base + HF_]; }
      else { fr = (1.f - a) * (float)Xf[base]; fi = (1.f - a) * (float)Xf[base + HF_]; }
    }
    Xfused[base] = (_Float16)fr;
    Xfused[base + HF_] = (_Float16)fi;
  }
}

// ---------------- depthwise 3x3 conv + add x_spec ----------------
__global__ __launch_bounds__(128) void k_conv_add(
    const _Float16* __restrict__ x2h, const _Float16* __restrict__ xspec,
    const float* __restrict__ convw, float* __restrict__ ssum) {
  __shared__ float tile[6][130];
  int bc = blockIdx.y;
  int y0 = blockIdx.x * 4;
  int tx = threadIdx.x;
  const _Float16* img = x2h + (size_t)bc * HW_;
#pragma unroll
  for (int i = 0; i < 6; ++i) {
    int y = y0 - 1 + i;
    tile[i][tx + 1] = (y >= 0 && y < 128) ? (float)img[y * 128 + tx] : 0.f;
  }
  if (tx == 0) {
#pragma unroll
    for (int i = 0; i < 6; ++i) { tile[i][0] = 0.f; tile[i][129] = 0.f; }
  }
  __syncthreads();
  int c = bc & (C_ - 1);
  float w9[9];
#pragma unroll
  for (int j = 0; j < 9; ++j) w9[j] = convw[c * 9 + j];
#pragma unroll
  for (int r = 0; r < 4; ++r) {
    float acc = 0.f;
#pragma unroll
    for (int i = 0; i < 3; ++i)
#pragma unroll
      for (int j = 0; j < 3; ++j)
        acc += w9[i * 3 + j] * tile[r + i][tx + j];
    size_t idx = (size_t)bc * HW_ + (size_t)(y0 + r) * 128 + tx;
    ssum[idx] = (float)xspec[idx] + acc;
  }
}

// ---------------- transpose back (b,c,p)->(b,p,c) with +lin ----------------
__global__ __launch_bounds__(256) void k_transpose_out(
    const float* __restrict__ ssum, const float* __restrict__ lin,
    float* __restrict__ pre) {
  __shared__ float t[32][33];
  int b = blockIdx.z;
  int c0 = blockIdx.x * 32, p0 = blockIdx.y * 32;
  int tx = threadIdx.x, ty = threadIdx.y;
#pragma unroll
  for (int i = 0; i < 4; ++i) {
    int c = c0 + ty + i * 8;
    t[ty + i * 8][tx] = ssum[((size_t)(b * C_ + c)) * HW_ + p0 + tx];
  }
  __syncthreads();
#pragma unroll
  for (int i = 0; i < 4; ++i) {
    int p = p0 + ty + i * 8;
    size_t o = ((size_t)b * HW_ + p) * C_ + c0 + tx;
    pre[o] = t[tx][ty + i * 8] + lin[o];
  }
}

// ---------------- conditional LayerNorm ----------------
__global__ __launch_bounds__(256) void k_ln(
    const float* __restrict__ pre, const float* __restrict__ timev,
    const float* __restrict__ ww, const float* __restrict__ wb,
    const float* __restrict__ bw, const float* __restrict__ bb,
    float* __restrict__ out) {
  int row = blockIdx.x;  // b*HW + p
  int c = threadIdx.x;
  float v = pre[(size_t)row * C_ + c];
  __shared__ float s1[256], s2[256];
  s1[c] = v;
  s2[c] = v * v;
  __syncthreads();
  for (int off = 128; off > 0; off >>= 1) {
    if (c < off) { s1[c] += s1[c + off]; s2[c] += s2[c + off]; }
    __syncthreads();
  }
  float mean = s1[0] * (1.f / 256.f);
  float var = s2[0] * (1.f / 256.f) - mean * mean;
  float inv = rsqrtf(var + 1e-5f);
  int b = row >> 14;
  float tt = timev[b];
  float w = tt * ww[c] + wb[c];
  float bi = tt * bw[c] + bb[c];
  out[(size_t)row * C_ + c] = w * (v - mean) * inv + bi;
}

// ---------------- host-side launch ----------------
extern "C" void kernel_launch(void* const* d_in, const int* in_sizes, int n_in,
                              void* d_out, int out_size, void* d_ws, size_t ws_size,
                              hipStream_t stream) {
  (void)in_sizes; (void)n_in; (void)out_size; (void)ws_size;
  const float* x      = (const float*)d_in[0];
  const float* timev  = (const float*)d_in[1];
  const float* w_real = (const float*)d_in[2];
  const float* w_imag = (const float*)d_in[3];
  const float* conv_w = (const float*)d_in[4];
  const float* lin_w  = (const float*)d_in[5];
  const float* lin_b  = (const float*)d_in[6];
  const float* mlp_w1 = (const float*)d_in[7];
  const float* mlp_b1 = (const float*)d_in[8];
  const float* mlp_w2 = (const float*)d_in[9];
  const float* mlp_b2 = (const float*)d_in[10];
  const float* n_ww   = (const float*)d_in[11];
  const float* n_wb   = (const float*)d_in[12];
  const float* n_bw   = (const float*)d_in[13];
  const float* n_bb   = (const float*)d_in[14];
  const float* kx     = (const float*)d_in[15];
  const float* ky     = (const float*)d_in[16];

  char* wsb = (char*)d_ws;
  _Float16* x2h   = (_Float16*)(wsb + O_X2H);
  _Float16* xh    = (_Float16*)(wsb + O_XH);
  _Float16* s1h   = (_Float16*)(wsb + O_S1);   // XrowRI -> Xfused
  _Float16* s2h   = (_Float16*)(wsb + O_S2);   // Xf -> xspec
  _Float16* s3h   = (_Float16*)(wsb + O_S3);   // Xmap -> Z
  float*    ssum  = (float*)(wsb + O_F1);
  float*    lin   = (float*)(wsb + O_F2);
  float*    pre   = (float*)(wsb + O_F3);
  _Float16* BfwdT = (_Float16*)(wsb + O_BFWD);
  _Float16* BinvT = (_Float16*)(wsb + O_BINV);
  _Float16* DrH   = (_Float16*)(wsb + O_DRH);
  _Float16* DiH   = (_Float16*)(wsb + O_DIH);
  _Float16* WrH   = (_Float16*)(wsb + O_WRH);
  _Float16* WiH   = (_Float16*)(wsb + O_WIH);
  _Float16* linT  = (_Float16*)(wsb + O_LNT);
  float*    cnt   = (float*)(wsb + O_CNT);
  float*    mean  = (float*)(wsb + O_MEAN);
  float*    alph  = (float*)(wsb + O_ALPH);

  k_init_consts<<<64, 256, 0, stream>>>(BfwdT, BinvT, DrH, DiH);
  k_init_counts<<<1, 256, 0, stream>>>(cnt);

  // f16 conversions of GEMM operands
  k_f32_to_f16<<<64, 256, 0, stream>>>(w_real, WrH, C_ * C_);
  k_f32_to_f16<<<64, 256, 0, stream>>>(w_imag, WiH, C_ * C_);
  k_f32_to_f16<<<64, 256, 0, stream>>>(lin_w, linT, C_ * C_);   // (N x K) already
  k_f32_to_f16<<<1024, 256, 0, stream>>>(x, xh, B_ * HW_ * C_);

  // (B,HW,C) -> (B*C, HW) in f16
  k_transpose_in<<<dim3(8, 512, 4), dim3(32, 8), 0, stream>>>(x, x2h);

  // row DFT: (131072 x 128) * BfwdT^T -> XrowRI (N=192 exact tiles)
  k_gemm<_Float16><<<dim3(3, 2048), 128, 0, stream>>>(
      x2h, 128, BfwdT, 128, nullptr, s1h, LDF_, 128);

  // column DFT (forward)
  k_coldft<<<dim3(3, 2, B_ * C_), 256, 0, stream>>>(DrH, DiH, s1h, s2h, +1);

  // band features -> alpha
  k_band_reduce<<<B_ * C_, 256, 0, stream>>>(s2h, cnt, mean);
  k_band_mlp<<<B_ * C_, 128, 0, stream>>>(mean, mlp_w1, mlp_b1, mlp_w2, mlp_b2, alph);

  // complex channel mix over masked low band
  k_chanmix<<<dim3(384, 4, B_), 256, 0, stream>>>(WrH, WiH, s2h, kx, ky, s3h);

  // fuse spectra
  k_fuse<<<B_ * C_, 256, 0, stream>>>(s2h, s3h, alph, kx, ky, s1h);

  // inverse column DFT (conjugate)
  k_coldft<<<dim3(3, 2, B_ * C_), 256, 0, stream>>>(DrH, DiH, s1h, s3h, -1);

  // inverse row (Hermitian fold): (131072 x 192) * BinvT^T -> x_spec (f16)
  k_gemm<_Float16><<<dim3(2, 2048), 128, 0, stream>>>(
      s3h, LDF_, BinvT, LDF_, nullptr, s2h, 128, LDF_);

  // depthwise conv + x_spec -> ssum (f32)
  k_conv_add<<<dim3(32, B_ * C_), 128, 0, stream>>>(x2h, s2h, conv_w, ssum);

  // pointwise linear: (65536 x 256) * lin_w^T + lin_b -> lin (f32)
  k_gemm<float><<<dim3(4, 1024), 128, 0, stream>>>(
      xh, C_, linT, C_, lin_b, lin, C_, C_);

  // (b,c,p) -> (b,p,c) with +lin
  k_transpose_out<<<dim3(8, 512, 4), dim3(32, 8), 0, stream>>>(ssum, lin, pre);

  // conditional LayerNorm -> d_out
  k_ln<<<B_ * HW_, 256, 0, stream>>>(pre, timev, n_ww, n_wb, n_bw, n_bb, (float*)d_out);
}
